// DeepComModel_55499567399442
// MI455X (gfx1250) — compile-verified, matrix operand order checked by reading, not run on hardware
//
#include <hip/hip_runtime.h>
#include <hip/hip_bf16.h>

// ---------------- types & helpers ----------------
typedef __attribute__((ext_vector_type(16))) __bf16 v16bf;
typedef __attribute__((ext_vector_type(8)))  __bf16 v8bf;
typedef __attribute__((ext_vector_type(8)))  float  v8f;

union BFrag { v16bf v; v8bf h[2]; };

__device__ __forceinline__ __bf16 f2bf(float f) {
    unsigned u = __builtin_bit_cast(unsigned, f);
    u += 0x7FFFu + ((u >> 16) & 1u);           // round-to-nearest-even
    unsigned short s = (unsigned short)(u >> 16);
    return __builtin_bit_cast(__bf16, s);
}
__device__ __forceinline__ float bf2f(__bf16 b) {
    unsigned short s = __builtin_bit_cast(unsigned short, b);
    unsigned u = ((unsigned)s) << 16;
    return __builtin_bit_cast(float, u);
}
__device__ __forceinline__ float sigf(float x) { return 1.f / (1.f + expf(-x)); }

__device__ __forceinline__ v16bf ldfrag(const __bf16* p) {
    BFrag f;
    f.h[0] = *reinterpret_cast<const v8bf*>(p);
    f.h[1] = *reinterpret_cast<const v8bf*>(p + 16);
    return f.v;
}
__device__ __forceinline__ v8f wmma_bf16(v16bf a, v16bf b, v8f c) {
    return __builtin_amdgcn_wmma_f32_16x16x32_bf16(false, a, false, b, (short)0, c, false, false);
}

#define HH 512
#define GDIM 1536
#define TT 500
#define BB 128
#define SUMV 30000
#define DECSTEPS 29

// ---------------- generic bf16 WMMA GEMM:  C[M,N] = A[M,K] @ W[N,K]^T + bias ----------------
// flags bit0: write bf16 (else f32); bit1: tanh activation
// Block: 256 thr = 8 waves arranged 2(M)x4(N); wave computes 16x64 (A-frag reused 4x).
// Wave id via readfirstlane so all tile predicates are SGPR-uniform (scalar branches only;
// EXEC stays all-ones for WMMA). Fast path is software-pipelined: fragments for k+32 are
// in flight while the 4 WMMAs for k execute.
__global__ void k_gemm_bf16(const __bf16* __restrict__ A, const __bf16* __restrict__ W,
                            const float* __restrict__ bias, void* __restrict__ Cout,
                            long ldC, int M, int N, int K, int flags) {
    const int lane = threadIdx.x & 31;
    const int wv   = __builtin_amdgcn_readfirstlane(threadIdx.x) >> 5;   // uniform
    const int mi = wv >> 2, ni = wv & 3;
    const int m0 = blockIdx.y * 32 + mi * 16;
    const int n0 = blockIdx.x * 256 + ni * 64;
    if (m0 >= M || n0 >= N) return;          // scalar branch
    const int lm = lane & 15, kh = lane >> 4;

    // number of active 16-wide tiles in this wave's 64-wide strip (N % 16 == 0 always here)
    const int nt = min(4, (N - n0) >> 4);    // uniform

    v8f acc0 = {}, acc1 = {}, acc2 = {}, acc3 = {};
    const __bf16* arow = A + (long)(m0 + lm) * K + 8 * kh;
    const __bf16* w0 = W + (long)(n0 +  0 + lm) * K + 8 * kh;
    const __bf16* w1 = W + (long)(min(n0 + 16, N - 16) + lm) * K + 8 * kh;
    const __bf16* w2 = W + (long)(min(n0 + 32, N - 16) + lm) * K + 8 * kh;
    const __bf16* w3 = W + (long)(min(n0 + 48, N - 16) + lm) * K + 8 * kh;

    if (nt == 4) {
        // software-pipelined fast path
        v16bf a  = ldfrag(arow);
        v16bf b0 = ldfrag(w0);
        v16bf b1 = ldfrag(w1);
        v16bf b2 = ldfrag(w2);
        v16bf b3 = ldfrag(w3);
        #pragma unroll 2
        for (int k = 32; k < K; k += 32) {
            v16bf an_ = ldfrag(arow + k);
            v16bf b0n = ldfrag(w0 + k);
            v16bf b1n = ldfrag(w1 + k);
            v16bf b2n = ldfrag(w2 + k);
            v16bf b3n = ldfrag(w3 + k);
            acc0 = wmma_bf16(a, b0, acc0);
            acc1 = wmma_bf16(a, b1, acc1);
            acc2 = wmma_bf16(a, b2, acc2);
            acc3 = wmma_bf16(a, b3, acc3);
            a = an_; b0 = b0n; b1 = b1n; b2 = b2n; b3 = b3n;
        }
        acc0 = wmma_bf16(a, b0, acc0);
        acc1 = wmma_bf16(a, b1, acc1);
        acc2 = wmma_bf16(a, b2, acc2);
        acc3 = wmma_bf16(a, b3, acc3);
    } else {
        // tail path (only the last N-block of wide GEMMs); nt uniform -> scalar branches
        for (int k = 0; k < K; k += 32) {
            v16bf a = ldfrag(arow + k);
            acc0 = wmma_bf16(a, ldfrag(w0 + k), acc0);
            if (nt > 1) acc1 = wmma_bf16(a, ldfrag(w1 + k), acc1);
            if (nt > 2) acc2 = wmma_bf16(a, ldfrag(w2 + k), acc2);
        }
    }

    v8f accs[4] = {acc0, acc1, acc2, acc3};
    #pragma unroll
    for (int q = 0; q < 4; ++q) {
        if (q >= nt) break;                  // uniform
        int col = n0 + 16 * q + lm;
        float bv = bias ? bias[col] : 0.f;
        #pragma unroll
        for (int r = 0; r < 8; ++r) {
            int m = m0 + kh * 8 + r;
            float v = accs[q][r] + bv;
            if (flags & 2) v = tanhf(v);
            long idx = (long)m * ldC + col;
            if (flags & 1) ((__bf16*)Cout)[idx] = f2bf(v);
            else           ((float*)Cout)[idx]  = v;
        }
    }
}

// ---------------- persistent recurrent GRU layer ----------------
// gix = x@Wi^T + bi precomputed (bf16, [T*B][1536]).  Block bc owns hidden cols J=16*bc.
// Wh slices (3 gates x 16 cols x 512 K) cached in LDS; A-fragments (h) pipelined from
// global; next timestep's gi rows prefetched (global_prefetch_b8) during gate math.
// Grid barrier (device-scope atomic + s_sleep spin) between timesteps.
__global__ void k_gru_layer(const __bf16* __restrict__ gix, const __bf16* __restrict__ Wh,
                            const float* __restrict__ bh,
                            __bf16* __restrict__ hA, __bf16* __restrict__ hB,
                            __bf16* __restrict__ ys, int outBmajor,
                            unsigned* __restrict__ counter) {
    __shared__ __align__(16) __bf16 sB[3][16][HH];
    const int J = blockIdx.x * 16;
    for (int i = threadIdx.x; i < 3 * 16 * HH; i += blockDim.x) {
        int k = i & (HH - 1); int n = (i >> 9) & 15; int g = i >> 13;
        sB[g][n][k] = Wh[(long)(g * HH + J + n) * HH + k];
    }
    __syncthreads();

    const int lane = threadIdx.x & 31;
    const int wv = __builtin_amdgcn_readfirstlane(threadIdx.x) >> 5;     // uniform
    const int lm = lane & 15, kh = lane >> 4;
    const int m0 = wv * 16;
    const int col = J + lm;
    const float bhr = bh[col], bhz = bh[512 + col], bhn = bh[1024 + col];

    __bf16* hc = hA; __bf16* hn = hB;
    volatile unsigned* vc = counter;

    for (int t = 0; t < TT; ++t) {
        v8f ar = {}, az = {}, an = {};
        const __bf16* arow = hc + (long)(m0 + lm) * HH + 8 * kh;
        v16bf a = ldfrag(arow);
        #pragma unroll
        for (int k = 0; k < HH; k += 32) {
            v16bf anx = a;
            if (k + 32 < HH) anx = ldfrag(arow + k + 32);   // compile-time guard (full unroll)
            v16bf b0 = ldfrag(&sB[0][lm][k + 8 * kh]);
            v16bf b1 = ldfrag(&sB[1][lm][k + 8 * kh]);
            v16bf b2 = ldfrag(&sB[2][lm][k + 8 * kh]);
            ar = wmma_bf16(a, b0, ar);
            az = wmma_bf16(a, b1, az);
            an = wmma_bf16(a, b2, an);
            a = anx;
        }
        // prefetch next timestep's gi rows for this lane's (m,col) positions
        if (t + 1 < TT) {
            const __bf16* gnext = gix + (long)((t + 1) * BB + m0 + kh * 8) * GDIM;
            __builtin_prefetch((const void*)(gnext + col), 0, 1);
            __builtin_prefetch((const void*)(gnext + 512 + col), 0, 1);
            __builtin_prefetch((const void*)(gnext + 1024 + col), 0, 1);
        }
        #pragma unroll
        for (int r = 0; r < 8; ++r) {
            int m = m0 + kh * 8 + r;
            const __bf16* gi = gix + (long)(t * BB + m) * GDIM;
            float ir = bf2f(gi[col]), iz = bf2f(gi[512 + col]), inn = bf2f(gi[1024 + col]);
            float rg = sigf(ir + ar[r] + bhr);
            float zg = sigf(iz + az[r] + bhz);
            float ng = tanhf(inn + rg * (an[r] + bhn));
            float hp = bf2f(hc[(long)m * HH + col]);
            float hv = (1.f - zg) * ng + zg * hp;
            __bf16 hb = f2bf(hv);
            hn[(long)m * HH + col] = hb;
            long orow = outBmajor ? ((long)m * TT + t) : ((long)t * BB + m);
            ys[orow * HH + col] = hb;
        }
        __threadfence();
        __syncthreads();
        if (threadIdx.x == 0) {
            atomicAdd(counter, 1u);
            unsigned tgt = (unsigned)(t + 1) * gridDim.x;
            while (*vc < tgt) { __builtin_amdgcn_s_sleep(2); }
        }
        __syncthreads();
        __threadfence();
        __bf16* tmp = hc; hc = hn; hn = tmp;
    }
}

// ---------------- small helper kernels ----------------
__global__ void k_f32_to_bf16(const float* __restrict__ in, __bf16* __restrict__ out, long n) {
    long i = (long)blockIdx.x * 256 + threadIdx.x;
    if (i < n) out[i] = f2bf(in[i]);
}
__global__ void k_enc_embed(const int* __restrict__ sbt, const float* __restrict__ emb,
                            __bf16* __restrict__ x) {
    long i = (long)blockIdx.x * 256 + threadIdx.x;     // (t*128+b)*256+d
    if (i >= (long)TT * BB * 256) return;
    int d = (int)(i & 255); long bt = i >> 8;
    int b = (int)(bt & 127); int t = (int)(bt >> 7);
    int tk = sbt[b * TT + t];
    x[i] = f2bf(emb[(long)tk * 256 + d]);
}
__global__ void k_dec_embed(const int* __restrict__ tok, const float* __restrict__ emb,
                            __bf16* __restrict__ out) {
    int i = blockIdx.x * 256 + threadIdx.x;            // 128*256
    int b = i >> 8, d = i & 255;
    out[i] = f2bf(emb[(long)tok[b] * 256 + d]);
}
__global__ void k_dec_gates(const float* __restrict__ gi, const float* __restrict__ gh,
                            __bf16* __restrict__ h) {
    int i = blockIdx.x * 256 + threadIdx.x;            // 128*512
    int b = i >> 9, j = i & 511;
    const float* gib = gi + (long)b * GDIM;
    const float* ghb = gh + (long)b * GDIM;
    float rg = sigf(gib[j] + ghb[j]);
    float zg = sigf(gib[512 + j] + ghb[512 + j]);
    float ng = tanhf(gib[1024 + j] + rg * ghb[1024 + j]);
    float hp = bf2f(h[i]);
    h[i] = f2bf((1.f - zg) * ng + zg * hp);
}
__global__ void k_attention(const __bf16* __restrict__ h, const __bf16* __restrict__ enc,
                            __bf16* __restrict__ ctx) {
    __shared__ float sh[HH];
    __shared__ float sc[TT];
    __shared__ float red[256];
    int b = blockIdx.x;
    for (int j = threadIdx.x; j < HH; j += 256) sh[j] = bf2f(h[b * HH + j]);
    __syncthreads();
    for (int t = threadIdx.x; t < TT; t += 256) {
        const __bf16* e = enc + ((long)b * TT + t) * HH;
        float acc = 0.f;
        for (int j = 0; j < HH; j += 8) {
            v8bf ev = *reinterpret_cast<const v8bf*>(e + j);
            #pragma unroll
            for (int q = 0; q < 8; ++q) acc += sh[j + q] * bf2f(ev[q]);
        }
        sc[t] = acc;
    }
    __syncthreads();
    float mx = -1e30f;
    for (int t = threadIdx.x; t < TT; t += 256) mx = fmaxf(mx, sc[t]);
    red[threadIdx.x] = mx; __syncthreads();
    for (int s = 128; s > 0; s >>= 1) {
        if (threadIdx.x < s) red[threadIdx.x] = fmaxf(red[threadIdx.x], red[threadIdx.x + s]);
        __syncthreads();
    }
    mx = red[0]; __syncthreads();
    float sm = 0.f;
    for (int t = threadIdx.x; t < TT; t += 256) { float e = expf(sc[t] - mx); sc[t] = e; sm += e; }
    red[threadIdx.x] = sm; __syncthreads();
    for (int s = 128; s > 0; s >>= 1) {
        if (threadIdx.x < s) red[threadIdx.x] += red[threadIdx.x + s];
        __syncthreads();
    }
    float inv = 1.f / red[0];
    int j0 = threadIdx.x, j1 = threadIdx.x + 256;
    float a0 = 0.f, a1 = 0.f;
    for (int t = 0; t < TT; ++t) {
        float p = sc[t] * inv;
        const __bf16* e = enc + ((long)b * TT + t) * HH;
        a0 += p * bf2f(e[j0]);
        a1 += p * bf2f(e[j1]);
    }
    ctx[b * HH + j0] = f2bf(a0);
    ctx[b * HH + j1] = f2bf(a1);
}
__global__ void k_argmax(const float* __restrict__ logits, long ld, int* __restrict__ tok) {
    __shared__ float bv[256]; __shared__ int bi[256];
    int b = blockIdx.x;
    const float* row = logits + (long)b * ld;
    float best = -1e30f; int bidx = 0;
    for (int v = threadIdx.x; v < SUMV; v += 256) {
        float x = row[v];
        if (x > best) { best = x; bidx = v; }
    }
    bv[threadIdx.x] = best; bi[threadIdx.x] = bidx; __syncthreads();
    for (int s = 128; s > 0; s >>= 1) {
        if (threadIdx.x < s) {
            if (bv[threadIdx.x + s] > bv[threadIdx.x] ||
                (bv[threadIdx.x + s] == bv[threadIdx.x] && bi[threadIdx.x + s] < bi[threadIdx.x])) {
                bv[threadIdx.x] = bv[threadIdx.x + s];
                bi[threadIdx.x] = bi[threadIdx.x + s];
            }
        }
        __syncthreads();
    }
    if (threadIdx.x == 0) tok[b] = bi[0];
}
__global__ void k_zero_u32(unsigned* __restrict__ p, int n) {
    int i = blockIdx.x * 256 + threadIdx.x;
    if (i < n) p[i] = 0u;
}
__global__ void k_set_tok(int* __restrict__ tok) {
    if (threadIdx.x < BB) tok[threadIdx.x] = 1;   // SOS
}

// ---------------- host orchestration ----------------
extern "C" void kernel_launch(void* const* d_in, const int* in_sizes, int n_in,
                              void* d_out, int out_size, void* d_ws, size_t ws_size,
                              hipStream_t stream) {
    (void)in_sizes; (void)n_in; (void)out_size; (void)ws_size;
    const int*   sbt      = (const int*)  d_in[0];
    const float* sbt_emb  = (const float*)d_in[1];
    const float* eWi0     = (const float*)d_in[2];
    const float* eWh0     = (const float*)d_in[3];
    const float* ebi0     = (const float*)d_in[4];
    const float* ebh0     = (const float*)d_in[5];
    const float* eWi1     = (const float*)d_in[6];
    const float* eWh1     = (const float*)d_in[7];
    const float* ebi1     = (const float*)d_in[8];
    const float* ebh1     = (const float*)d_in[9];
    const float* sum_emb  = (const float*)d_in[10];
    const float* dWi      = (const float*)d_in[11];
    const float* dWh      = (const float*)d_in[12];
    const float* dbi      = (const float*)d_in[13];
    const float* dbh      = (const float*)d_in[14];
    const float* pW1      = (const float*)d_in[15];
    const float* pb1      = (const float*)d_in[16];
    const float* pW2      = (const float*)d_in[17];
    const float* pb2      = (const float*)d_in[18];
    float* out = (float*)d_out;

    // workspace layout (256B aligned)
    size_t off = 0;
    auto alloc = [&](size_t bytes) { size_t r = off; off = (off + bytes + 255) & ~(size_t)255; return r; };
    char* ws = (char*)d_ws;
    auto BF = [&](size_t o) { return (__bf16*)(ws + o); };
    auto F32 = [&](size_t o) { return (float*)(ws + o); };

    size_t oWi0 = alloc(1536u * 256 * 2);
    size_t oWh0 = alloc(1536u * 512 * 2);
    size_t oWi1 = alloc(1536u * 512 * 2);
    size_t oWh1 = alloc(1536u * 512 * 2);
    size_t oDWi = alloc(1536u * 256 * 2);
    size_t oDWh = alloc(1536u * 512 * 2);
    size_t oPW1 = alloc(1536u * 512 * 2);
    size_t oPW2 = alloc((size_t)SUMV * 1536 * 2);
    size_t oX0  = alloc((size_t)TT * BB * 256 * 2);
    size_t oGIX = alloc((size_t)TT * BB * GDIM * 2);
    size_t oYS0 = alloc((size_t)TT * BB * HH * 2);
    size_t oENC = alloc((size_t)BB * TT * HH * 2);
    size_t oHA  = alloc((size_t)BB * HH * 2);
    size_t oHB  = alloc((size_t)BB * HH * 2);
    size_t oCTR = alloc(256);
    size_t oEMB = alloc((size_t)BB * 256 * 2);
    size_t oGID = alloc((size_t)BB * GDIM * 4);
    size_t oGHD = alloc((size_t)BB * GDIM * 4);
    size_t oHD  = alloc((size_t)BB * HH * 2);
    size_t oCTX = alloc((size_t)BB * HH * 2);
    size_t oU   = alloc((size_t)BB * GDIM * 2);
    size_t oTOK = alloc((size_t)BB * 4);

    auto conv = [&](const float* src, size_t dsto, long n) {
        k_f32_to_bf16<<<(unsigned)((n + 255) / 256), 256, 0, stream>>>(src, BF(dsto), n);
    };
    auto gemm = [&](const __bf16* A, const __bf16* W, const float* bias, void* C, long ldC,
                    int M, int N, int K, int flags) {
        dim3 g((N + 255) / 256, (M + 31) / 32);
        k_gemm_bf16<<<g, 256, 0, stream>>>(A, W, bias, C, ldC, M, N, K, flags);
    };
    auto zero = [&](size_t o, int nwords) {
        k_zero_u32<<<(nwords + 255) / 256, 256, 0, stream>>>((unsigned*)(ws + o), nwords);
    };

    // ---- weight conversion to bf16 ----
    conv(eWi0, oWi0, 1536L * 256);
    conv(eWh0, oWh0, 1536L * 512);
    conv(eWi1, oWi1, 1536L * 512);
    conv(eWh1, oWh1, 1536L * 512);
    conv(dWi,  oDWi, 1536L * 256);
    conv(dWh,  oDWh, 1536L * 512);
    conv(pW1,  oPW1, 1536L * 512);
    conv(pW2,  oPW2, (long)SUMV * 1536);

    // ---- encoder ----
    long nx = (long)TT * BB * 256;
    k_enc_embed<<<(unsigned)((nx + 255) / 256), 256, 0, stream>>>(sbt, sbt_emb, BF(oX0));
    // gix0 = x @ Wi0^T + bi0  (bulk GEMM, no serial dependency)
    gemm(BF(oX0), BF(oWi0), ebi0, BF(oGIX), GDIM, TT * BB, GDIM, 256, /*bf16 out*/1);
    zero(oHA, BB * HH / 2); zero(oHB, BB * HH / 2); zero(oCTR, 1);
    k_gru_layer<<<32, 256, 0, stream>>>(BF(oGIX), BF(oWh0), ebh0, BF(oHA), BF(oHB),
                                        BF(oYS0), /*Tmajor*/0, (unsigned*)(ws + oCTR));
    // gix1 = ys0 @ Wi1^T + bi1
    gemm(BF(oYS0), BF(oWi1), ebi1, BF(oGIX), GDIM, TT * BB, GDIM, 512, 1);
    zero(oHA, BB * HH / 2); zero(oHB, BB * HH / 2); zero(oCTR, 1);
    k_gru_layer<<<32, 256, 0, stream>>>(BF(oGIX), BF(oWh1), ebh1, BF(oHA), BF(oHB),
                                        BF(oENC), /*Bmajor*/1, (unsigned*)(ws + oCTR));

    // ---- decoder: 29 greedy steps, stream-ordered ----
    zero(oHD, BB * HH / 2);
    k_set_tok<<<1, 128, 0, stream>>>((int*)(ws + oTOK));
    const long ldOut = (long)DECSTEPS * SUMV;
    for (int s = 0; s < DECSTEPS; ++s) {
        k_dec_embed<<<BB, 256, 0, stream>>>((int*)(ws + oTOK), sum_emb, BF(oEMB));
        gemm(BF(oEMB), BF(oDWi), dbi, F32(oGID), GDIM, BB, GDIM, 256, 0);
        gemm(BF(oHD),  BF(oDWh), dbh, F32(oGHD), GDIM, BB, GDIM, 512, 0);
        k_dec_gates<<<(BB * HH) / 256, 256, 0, stream>>>(F32(oGID), F32(oGHD), BF(oHD));
        k_attention<<<BB, 256, 0, stream>>>(BF(oHD), BF(oENC), BF(oCTX));
        gemm(BF(oCTX), BF(oPW1), pb1, BF(oU), GDIM, BB, GDIM, 512, /*bf16|tanh*/3);
        gemm(BF(oU), BF(oPW2), pb2, out + (long)s * SUMV, ldOut, BB, SUMV, GDIM, 0);
        k_argmax<<<BB, 256, 0, stream>>>(out + (long)s * SUMV, ldOut, (int*)(ws + oTOK));
    }
}
